// GDAttention_89928025244464
// MI455X (gfx1250) — matrix operand order repaired and checked by reading, not compiled, not run on hardware
//
#include <hip/hip_runtime.h>

#define S_LEN 2048
#define D_DIM 256
#define B_N   8
#define H_N   8

typedef _Float16 v16h __attribute__((ext_vector_type(16)));
typedef _Float16 v8h  __attribute__((ext_vector_type(8)));
typedef float    v8f  __attribute__((ext_vector_type(8)));

#define KSTRIDE 264              // halves per K-row (256 + pad, keeps 16B align)
#define ESTRIDE 40               // halves per transposed-E row (32 + pad)
#define YSTRIDE 264              // halves per Q/y row (region is dual-purpose)
#define SMEM_HALVES (32*KSTRIDE + 256*ESTRIDE + 8*16*32 + 8*16*YSTRIDE)
#define SMEM_BYTES  (SMEM_HALVES * 2)

// Fold W_o (256 x 2048) into W_sum (256 x 256) f16: W_sum[d][k] = sum_h W_o[d][h*256+k]
__global__ void wsum_prep_kernel(const float* __restrict__ Wo, _Float16* __restrict__ Wf) {
    int idx = blockIdx.x * 256 + threadIdx.x;      // 0 .. 65535
    int dn  = idx >> 8;
    int k   = idx & 255;
    float s = 0.f;
#pragma unroll
    for (int h = 0; h < H_N; ++h) s += Wo[(size_t)dn * (D_DIM * H_N) + h * D_DIM + k];
    Wf[idx] = (_Float16)s;
}

// Build a 16-half fragment from two aligned 8-half chunks (LDS or global).
__device__ __forceinline__ v16h ld16h(const _Float16* ptr) {
    const v8h lo = *(const v8h*)ptr;
    const v8h hi = *(const v8h*)(ptr + 8);
    v16h r;
#pragma unroll
    for (int j = 0; j < 8; ++j) { r[j] = lo[j]; r[8 + j] = hi[j]; }
    return r;
}

// A-fragment (16x32 f16): lane row m = lane&15; element j -> k = 16*h + j.
// rowptr points at the row's chunk base (c*32); returns packed v16h.
__device__ __forceinline__ v16h ldAfrag(const _Float16* rowptr, int h) {
    const v8h lo = *(const v8h*)(rowptr + 16 * h);
    const v8h hi = *(const v8h*)(rowptr + 16 * h + 8);
    v16h a;
#pragma unroll
    for (int j = 0; j < 8; ++j) { a[j] = lo[j]; a[8 + j] = hi[j]; }
    return a;
}

__launch_bounds__(256, 1)
__global__ void gd_attn_kernel(const float* __restrict__ e,
                               const float* __restrict__ p,
                               const _Float16* __restrict__ Wf,
                               float* __restrict__ out) {
    extern __shared__ _Float16 smem[];
    _Float16* kbuf  = smem;                        // [32][KSTRIDE]  K tile, row-major f16
    _Float16* ebufT = kbuf  + 32 * KSTRIDE;        // [256][ESTRIDE] E tile, transposed f16
    _Float16* pbuf  = ebufT + 256 * ESTRIDE;       // [8 waves][16][32] P round-trip
    _Float16* qybuf = pbuf  + 8 * 16 * 32;         // [8 waves][16][YSTRIDE] Q, later y

    const int tid  = threadIdx.x;
    const int wave = tid >> 5;
    const int lane = tid & 31;
    const int n    = lane & 15;                    // B/C/D column within tile
    const int h    = lane >> 4;                    // half-wave select

    const int bq  = blockIdx.y;                    // batch
    const int q0b = blockIdx.x * 128;              // first query row of block
    const int q0w = q0b + wave * 16;               // first query row of this wave

    _Float16* qy = &qybuf[wave * 16 * YSTRIDE];    // wave-private Q/y region

    // ---- stage this wave's Q tile (16 x 256) into LDS as f16.
    // lane covers row (lane&15), column half h*128..h*128+127.
    {
        const float* qrow = p + ((size_t)bq * (S_LEN + 1) + (size_t)(q0w + n + 1)) * D_DIM + h * 128;
        _Float16* qdst = &qy[n * YSTRIDE + h * 128];
#pragma unroll
        for (int i = 0; i < 32; ++i) {
            const float4 q4 = *(const float4*)(qrow + i * 4);
            qdst[i * 4 + 0] = (_Float16)q4.x;
            qdst[i * 4 + 1] = (_Float16)q4.y;
            qdst[i * 4 + 2] = (_Float16)q4.z;
            qdst[i * 4 + 3] = (_Float16)q4.w;
        }
    }
    asm volatile("s_wait_dscnt 0x0" ::: "memory"); // Q visible wave-wide

    float mrun[8], lrun[8];
#pragma unroll
    for (int r = 0; r < 8; ++r) { mrun[r] = -1e30f; lrun[r] = 0.f; }
    v8f oacc[16];
#pragma unroll
    for (int dt = 0; dt < 16; ++dt)
#pragma unroll
        for (int r = 0; r < 8; ++r) oacc[dt][r] = 0.f;

    const _Float16* qArow = &qy[(lane & 15) * YSTRIDE];   // A-frag row base for this lane

    // prefetch coordinates: one 128B line of K-row and E-row per thread per step
    const int pfrow = tid >> 3;                    // 0..31
    const int pfcol = (tid & 7) << 5;              // 0..224 floats (128B chunks)

    const int kvmax = q0b + 127;
    for (int kv0 = 0; kv0 <= kvmax; kv0 += 32) {
        // ---- cooperative stage: K rows (p[:, kv0..kv0+31, :]) and E rows (transposed)
#pragma unroll
        for (int it = 0; it < 8; ++it) {
            int idx = tid + it * 256;              // 0 .. 2047
            int kvl = idx >> 6;                    // 0 .. 31
            int dv  = (idx & 63) << 2;             // 0 .. 252 step 4
            const float4 k4 = *(const float4*)(p + ((size_t)bq * (S_LEN + 1) + (size_t)(kv0 + kvl)) * D_DIM + dv);
            _Float16* kd = &kbuf[kvl * KSTRIDE + dv];
            kd[0] = (_Float16)k4.x; kd[1] = (_Float16)k4.y;
            kd[2] = (_Float16)k4.z; kd[3] = (_Float16)k4.w;
            const float4 e4 = *(const float4*)(e + ((size_t)bq * S_LEN + (size_t)(kv0 + kvl)) * D_DIM + dv);
            ebufT[(dv + 0) * ESTRIDE + kvl] = (_Float16)e4.x;
            ebufT[(dv + 1) * ESTRIDE + kvl] = (_Float16)e4.y;
            ebufT[(dv + 2) * ESTRIDE + kvl] = (_Float16)e4.z;
            ebufT[(dv + 3) * ESTRIDE + kvl] = (_Float16)e4.w;
        }
        __syncthreads();

        // ---- prefetch next iteration's K/E tiles; flights during the WMMA phase
        const int kvn = kv0 + 32;
        if (kvn <= kvmax) {
            __builtin_prefetch(p + ((size_t)bq * (S_LEN + 1) + (size_t)(kvn + pfrow)) * D_DIM + pfcol, 0, 3);
            __builtin_prefetch(e + ((size_t)bq * S_LEN + (size_t)(kvn + pfrow)) * D_DIM + pfcol, 0, 3);
        }

        if (kv0 <= q0w + 15) {                     // wave-uniform causal skip
            // ---- S = Q K^T for two 16-wide kv tiles (kv0..+15, kv0+16..+31)
            v8f s0, s1;
#pragma unroll
            for (int r = 0; r < 8; ++r) { s0[r] = 0.f; s1[r] = 0.f; }
#pragma unroll
            for (int c = 0; c < 8; ++c) {
                const v16h aq  = ldAfrag(qArow + c * 32, h);
                const v16h bk0 = ld16h(&kbuf[n * KSTRIDE + c * 32 + 16 * h]);
                s0 = __builtin_amdgcn_wmma_f32_16x16x32_f16(false, aq, false, bk0, (short)0, s0, false, false);
                const v16h bk1 = ld16h(&kbuf[(16 + n) * KSTRIDE + c * 32 + 16 * h]);
                s1 = __builtin_amdgcn_wmma_f32_16x16x32_f16(false, aq, false, bk1, (short)0, s1, false, false);
            }

            // ---- online softmax (rows r+8h live across the 16 lanes of each half)
            float alpha[8];
            _Float16* pb = &pbuf[wave * 512];
#pragma unroll
            for (int r = 0; r < 8; ++r) {
                const int qg = q0w + r + 8 * h;
                float sv0 = fminf(fmaxf(s0[r] * 0.0625f, -10.f), 10.f);
                if (kv0 + n > qg)      sv0 = -1e30f;
                float sv1 = fminf(fmaxf(s1[r] * 0.0625f, -10.f), 10.f);
                if (kv0 + 16 + n > qg) sv1 = -1e30f;
                float rmax = fmaxf(sv0, sv1);
                rmax = fmaxf(rmax, __shfl_xor(rmax, 1, 32));
                rmax = fmaxf(rmax, __shfl_xor(rmax, 2, 32));
                rmax = fmaxf(rmax, __shfl_xor(rmax, 4, 32));
                rmax = fmaxf(rmax, __shfl_xor(rmax, 8, 32));
                const float mnew = fmaxf(mrun[r], rmax);
                alpha[r] = __expf(mrun[r] - mnew);
                const float p0 = (sv0 > -1e29f) ? __expf(sv0 - mnew) : 0.f;
                const float p1 = (sv1 > -1e29f) ? __expf(sv1 - mnew) : 0.f;
                float rs = p0 + p1;
                rs += __shfl_xor(rs, 1, 32);
                rs += __shfl_xor(rs, 2, 32);
                rs += __shfl_xor(rs, 4, 32);
                rs += __shfl_xor(rs, 8, 32);
                lrun[r] = lrun[r] * alpha[r] + rs;
                mrun[r] = mnew;
                const int m = r + 8 * h;
                pb[m * 32 + n]      = (_Float16)p0;   // C-layout -> LDS
                pb[m * 32 + 16 + n] = (_Float16)p1;
            }
#pragma unroll
            for (int dt = 0; dt < 16; ++dt)
#pragma unroll
                for (int r = 0; r < 8; ++r) oacc[dt][r] *= alpha[r];

            asm volatile("s_wait_dscnt 0x0" ::: "memory");   // P stores visible wave-wide

            // ---- P (16x32) as A-fragment, then O += P @ E for all 16 d-column tiles
            const v16h pa = ldAfrag(&pbuf[wave * 512 + (lane & 15) * 32], h);
#pragma unroll
            for (int dt = 0; dt < 16; ++dt) {
                const v16h be = ld16h(&ebufT[(dt * 16 + n) * ESTRIDE + 16 * h]);
                oacc[dt] = __builtin_amdgcn_wmma_f32_16x16x32_f16(false, pa, false, be, (short)0, oacc[dt], false, false);
            }
        }
        __syncthreads();
    }

    // ---- normalize (1/l) and row_scale 1/(q+1); stash y as f16 over the Q region
    float oscale[8];
#pragma unroll
    for (int r = 0; r < 8; ++r) {
        const int qg = q0w + r + 8 * h;
        oscale[r] = 1.f / (lrun[r] * (float)(qg + 1));
    }
#pragma unroll
    for (int dt = 0; dt < 16; ++dt)
#pragma unroll
        for (int r = 0; r < 8; ++r)
            qy[(r + 8 * h) * YSTRIDE + dt * 16 + n] = (_Float16)(oacc[dt][r] * oscale[r]);
    asm volatile("s_wait_dscnt 0x0" ::: "memory");

    // ---- out = y @ W_sum^T  (W_sum row-major f16: B[k][ncol] contiguous per lane)
#pragma unroll
    for (int ncol = 0; ncol < 16; ++ncol) {
        v8f acc;
#pragma unroll
        for (int r = 0; r < 8; ++r) acc[r] = 0.f;
#pragma unroll
        for (int c = 0; c < 8; ++c) {
            const v16h ya = ldAfrag(qArow + c * 32, h);
            const v16h bw = ld16h(&Wf[(size_t)(ncol * 16 + n) * 256 + c * 32 + 16 * h]);
            acc = __builtin_amdgcn_wmma_f32_16x16x32_f16(false, ya, false, bw, (short)0, acc, false, false);
        }
#pragma unroll
        for (int r = 0; r < 8; ++r)
            out[((size_t)bq * S_LEN + (size_t)(q0w + r + 8 * h)) * D_DIM + ncol * 16 + n] = acc[r];
    }
}

extern "C" void kernel_launch(void* const* d_in, const int* in_sizes, int n_in,
                              void* d_out, int out_size, void* d_ws, size_t ws_size,
                              hipStream_t stream) {
    const float* e  = (const float*)d_in[0];
    const float* p  = (const float*)d_in[1];
    const float* Wo = (const float*)d_in[2];
    float* out = (float*)d_out;
    _Float16* Wf = (_Float16*)d_ws;                 // 256*256*2 = 128 KB scratch

    (void)hipFuncSetAttribute((const void*)gd_attn_kernel,
                              hipFuncAttributeMaxDynamicSharedMemorySize, SMEM_BYTES);

    wsum_prep_kernel<<<256, 256, 0, stream>>>(Wo, Wf);
    dim3 grid(S_LEN / 128, B_N);                    // 16 query-blocks x 8 batches
    gd_attn_kernel<<<grid, 256, SMEM_BYTES, stream>>>(e, p, Wf, out);
}
